// EquiMHA_36584531427724
// MI455X (gfx1250) — matrix operand order repaired and checked by reading, not compiled, not run on hardware
//
#include <hip/hip_runtime.h>
#include <hip/hip_bf16.h>
#include <math.h>

typedef __bf16 bf16;
typedef __attribute__((ext_vector_type(16))) __bf16 v16bf;
typedef __attribute__((ext_vector_type(8)))  __bf16 v8bf;
typedef __attribute__((ext_vector_type(4)))  __bf16 v4bf;
typedef __attribute__((ext_vector_type(8)))  float   v8f;
typedef __attribute__((ext_vector_type(4)))  int     v4i;

#define BB  8
#define NN  512
#define MM  4
#define KNN 512
#define DQK 512
#define DVV 1024
#define HH  16
#define DD  64
#define ROWS (BB*NN*MM)   // 16384

#define AS1 __attribute__((address_space(1)))
#define AS3 __attribute__((address_space(3)))

#if defined(__AMDGCN__) && __has_builtin(__builtin_amdgcn_global_load_async_to_lds_b128)
  #define HAVE_ASYNC_LDS 1
#else
  #define HAVE_ASYNC_LDS 0
#endif

__device__ __forceinline__ void wait_async_all() {
#if HAVE_ASYNC_LDS
  #if __has_builtin(__builtin_amdgcn_s_wait_asynccnt)
    __builtin_amdgcn_s_wait_asynccnt(0);
  #else
    asm volatile("s_wait_asynccnt 0x0" ::: "memory");
  #endif
#endif
}

__device__ __forceinline__ v8f wmma_bf16(v16bf a, v16bf b, v8f c) {
    return __builtin_amdgcn_wmma_f32_16x16x32_bf16(false, a, false, b, (short)0, c, false, false);
}

// ---------------------------------------------------------------------------
// 16-lane row reductions via DPP butterflies (no LDS traffic).
// After xor1+xor2 the quads are uniform, so ROW_HALF_MIRROR acts as xor4 and
// ROW_MIRROR as xor8 -> full reduction within each 16-lane half of the wave,
// which matches the WMMA C-fragment row halves.
// ---------------------------------------------------------------------------
#if defined(__AMDGCN__) && __has_builtin(__builtin_amdgcn_update_dpp)
template<int CTRL>
__device__ __forceinline__ float dpp_xfer(float v) {
    return __int_as_float(__builtin_amdgcn_update_dpp(
        0, __float_as_int(v), CTRL, 0xF, 0xF, true));
}
__device__ __forceinline__ float rowmax16(float v) {
    v = fmaxf(v, dpp_xfer<0xB1>(v));    // quad_perm [1,0,3,2]  (xor 1)
    v = fmaxf(v, dpp_xfer<0x4E>(v));    // quad_perm [2,3,0,1]  (xor 2)
    v = fmaxf(v, dpp_xfer<0x141>(v));   // row_half_mirror      (xor 4 once quads uniform)
    v = fmaxf(v, dpp_xfer<0x140>(v));   // row_mirror           (xor 8 once octs uniform)
    return v;
}
__device__ __forceinline__ float rowsum16(float v) {
    v += dpp_xfer<0xB1>(v);
    v += dpp_xfer<0x4E>(v);
    v += dpp_xfer<0x141>(v);
    v += dpp_xfer<0x140>(v);
    return v;
}
#else
__device__ __forceinline__ float rowmax16(float v) {
    for (int o = 8; o; o >>= 1) v = fmaxf(v, __shfl_xor(v, o, 32));
    return v;
}
__device__ __forceinline__ float rowsum16(float v) {
    for (int o = 8; o; o >>= 1) v += __shfl_xor(v, o, 32);
    return v;
}
#endif

// ---------------------------------------------------------------------------
// f32 -> bf16 elementwise convert (4 elems / thread)
// ---------------------------------------------------------------------------
__global__ void cvt_f32_bf16_kernel(const float* __restrict__ in, bf16* __restrict__ out, int n4) {
    int i = blockIdx.x * blockDim.x + threadIdx.x;
    if (i < n4) {
        float4 v = ((const float4*)in)[i];
        v4bf o;
        o[0] = (bf16)v.x; o[1] = (bf16)v.y; o[2] = (bf16)v.z; o[3] = (bf16)v.w;
        ((v4bf*)out)[i] = o;
    }
}

// ---------------------------------------------------------------------------
// f32 [R][C] -> bf16 transposed [C][R]
// ---------------------------------------------------------------------------
__global__ void transpose_cvt_kernel(const float* __restrict__ in, bf16* __restrict__ out,
                                     int R, int C) {
    int c = blockIdx.x * blockDim.x + threadIdx.x;
    int r = blockIdx.y;
    if (c < C) out[(size_t)c * R + r] = (bf16)in[(size_t)r * C + c];
}

// ---------------------------------------------------------------------------
// bf16 GEMM:  C[rows x ncols] = oscale * (A[rows x kdim] * Bt[ncols x kdim]^T)
//   MODE 0: store bf16 row-major     MODE 1: store f32 row-major
//   MODE 2: store bf16 into V-transposed layout [b][m][DVV][KNN]
// Block: 256 threads (8 waves). Wave tile: 16 rows x 64 cols.
// B tile (shared by all 8 waves) staged in LDS, double buffered, async copy.
// ---------------------------------------------------------------------------
template<int MODE>
__global__ void __launch_bounds__(256)
gemm_bf16_kernel(const bf16* __restrict__ A, const bf16* __restrict__ Bt,
                 void* __restrict__ Cout, int kdim, int ncols, float oscale) {
    __shared__ __align__(16) bf16 Bs[2][64][40];   // 40: pad rows to 80B

    const int tid  = threadIdx.x;
    const int lane = tid & 31;
    const int wave = tid >> 5;
    const int ln   = lane & 15;
    const int hf   = lane >> 4;
    const int row0 = blockIdx.x * 128 + wave * 16;
    const int col0 = blockIdx.y * 64;

    // staging assignment: thread -> (column, 8-elem k chunk), one b128 per thread
    const int scol = tid >> 2;          // 0..63
    const int skp  = (tid & 3) * 8;     // 0,8,16,24
    const bf16* sg = Bt + (size_t)(col0 + scol) * kdim + skp;

    auto stage = [&](int buf, const bf16* src) {
#if HAVE_ASYNC_LDS
        __builtin_amdgcn_global_load_async_to_lds_b128(
            (AS1 v4i*)(uintptr_t)src,
            (AS3 v4i*)(void*)&Bs[buf][scol][skp],
            0, 0);
#else
        *(v8bf*)&Bs[buf][scol][skp] = *(const v8bf*)src;
#endif
    };

    stage(0, sg);

    v8f acc[4] = {};
    const int steps = kdim >> 5;
    for (int i = 0; i < steps; ++i) {
        const int k0  = i << 5;
        const int cur = i & 1;

        wait_async_all();      // my async writes for buffer `cur` have landed
        __syncthreads();       // everyone's writes landed; prev-iter reads done
        if (i + 1 < steps) stage(cur ^ 1, sg + k0 + 32);

        // A fragment (16x32): lane = row M, halves split K (0..15 | 16..31)
        const bf16* ap = A + (size_t)(row0 + ln) * kdim + k0 + hf * 16;
        v8bf lo = *(const v8bf*)ap;
        v8bf hi = *(const v8bf*)(ap + 8);
        v16bf af;
#pragma unroll
        for (int e = 0; e < 8; ++e) { af[e] = lo[e]; af[e + 8] = hi[e]; }
        if (i + 1 < steps) __builtin_prefetch(ap + 32, 0, 1);

#pragma unroll
        for (int j = 0; j < 4; ++j) {
            // B fragment (32x16) from LDS: lane = column N, halves split K
            const bf16* bp = &Bs[cur][16 * j + ln][hf * 16];
            acc[j] = wmma_bf16(af, *(const v16bf*)bp, acc[j]);
        }
    }

#pragma unroll
    for (int j = 0; j < 4; ++j) {
        int n = col0 + 16 * j + ln;
#pragma unroll
        for (int r = 0; r < 8; ++r) {
            int row = row0 + r + hf * 8;          // C layout: vgpr r -> M=r (+8 upper half)
            float v = acc[j][r] * oscale;
            if (MODE == 0) {
                ((bf16*)Cout)[(size_t)row * ncols + n] = (bf16)v;
            } else if (MODE == 1) {
                ((float*)Cout)[(size_t)row * ncols + n] = v;
            } else {
                int m = row & 3, k = (row >> 2) & 511, b = row >> 11;
                ((bf16*)Cout)[(((size_t)b * MM + m) * DVV + n) * KNN + k] = (bf16)v;
            }
        }
    }
}

// ---------------------------------------------------------------------------
// Two-pass attention. 1 wave per (b, h, 16-query tile).
// Pass 1: row max over all 512 keys (reference takes max over ALL keys).
// Pass 2: p = mask ? exp(s - max) : 0 ; lsum += rowsum(p) ; O += P*V.
// No online rescaling -> no alpha multiply storm on the 128-VGPR accumulator.
// Qp (pre-scaled by 1/sqrt(DV)), Kp: bf16 [16384][1024].  VpT: bf16 [b][m][1024][512].
// ---------------------------------------------------------------------------
__global__ void __launch_bounds__(32)
attn_kernel(const bf16* __restrict__ Qp, const bf16* __restrict__ Kp,
            const bf16* __restrict__ VpT, const int* __restrict__ mask,
            bf16* __restrict__ Obf) {
    __shared__ __align__(16) bf16 P[16][40];

    const int lane = threadIdx.x & 31;
    const int ln = lane & 15, hf = lane >> 4;
    const int n0 = blockIdx.x * 16;
    const int h  = blockIdx.y;
    const int b  = blockIdx.z;

    // Preload 8 Q A-fragments covering contraction c = (m,d) in [0,256)
    v16bf qf[8];
#pragma unroll
    for (int f = 0; f < 8; ++f) {
        int m = f >> 1, dw = (f & 1) * 32;
        const bf16* qp = Qp + (((size_t)(b * NN + n0 + ln)) * MM + m) * DVV + h * DD + dw + hf * 16;
        v8bf lo = *(const v8bf*)qp;
        v8bf hi = *(const v8bf*)(qp + 8);
#pragma unroll
        for (int e = 0; e < 8; ++e) { qf[f][e] = lo[e]; qf[f][e + 8] = hi[e]; }
    }

    // ---------------- pass 1: global row max ----------------
    float rmax[8];
#pragma unroll
    for (int r = 0; r < 8; ++r) rmax[r] = -INFINITY;

    for (int key0 = 0; key0 < KNN; key0 += 32) {
        v16bf kf[16];
#pragma unroll
        for (int f = 0; f < 8; ++f) {
            int m = f >> 1, dw = (f & 1) * 32;
            const bf16* kp0 = Kp + (((size_t)(b * NN + key0 + ln)) * MM + m) * DVV + h * DD + dw + hf * 16;
            kf[f]     = *(const v16bf*)kp0;
            kf[f + 8] = *(const v16bf*)(kp0 + (size_t)16 * MM * DVV);   // +16 keys
        }
        v8f e0 = {}, e1 = {};
#pragma unroll
        for (int f = 0; f < 8; ++f) e0 = wmma_bf16(qf[f], kf[f], e0);
#pragma unroll
        for (int f = 0; f < 8; ++f) e1 = wmma_bf16(qf[f], kf[f + 8], e1);
#pragma unroll
        for (int r = 0; r < 8; ++r)
            rmax[r] = fmaxf(rmax[r], rowmax16(fmaxf(e0[r], e1[r])));
    }

    // ---------------- pass 2: probabilities + O accumulation ----------------
    float lsum[8];
#pragma unroll
    for (int r = 0; r < 8; ++r) lsum[r] = 0.f;
    v8f oacc[16] = {};

    for (int key0 = 0; key0 < KNN; key0 += 32) {
        v8f e0 = {}, e1 = {};
        // scores for keys key0..key0+15, fragments in groups of 4 (bounded reg pressure)
#pragma unroll
        for (int g = 0; g < 2; ++g) {
            v16bf kg[4];
#pragma unroll
            for (int t = 0; t < 4; ++t) {
                int f = g * 4 + t, m = f >> 1, dw = (f & 1) * 32;
                kg[t] = *(const v16bf*)(Kp + (((size_t)(b * NN + key0 + ln)) * MM + m) * DVV
                                        + h * DD + dw + hf * 16);
            }
#pragma unroll
            for (int t = 0; t < 4; ++t) e0 = wmma_bf16(qf[g * 4 + t], kg[t], e0);
        }
        // scores for keys key0+16..key0+31
#pragma unroll
        for (int g = 0; g < 2; ++g) {
            v16bf kg[4];
#pragma unroll
            for (int t = 0; t < 4; ++t) {
                int f = g * 4 + t, m = f >> 1, dw = (f & 1) * 32;
                kg[t] = *(const v16bf*)(Kp + (((size_t)(b * NN + key0 + 16 + ln)) * MM + m) * DVV
                                        + h * DD + dw + hf * 16);
            }
#pragma unroll
            for (int t = 0; t < 4; ++t) e1 = wmma_bf16(qf[g * 4 + t], kg[t], e1);
        }

#pragma unroll
        for (int r = 0; r < 8; ++r) {
            int nq = n0 + r + hf * 8;
            const int* mrow = mask + ((size_t)(b * NN) + nq) * KNN + key0 + ln;
            float p0 = mrow[0]  ? __expf(e0[r] - rmax[r]) : 0.f;
            float p1 = mrow[16] ? __expf(e1[r] - rmax[r]) : 0.f;
            lsum[r] += rowsum16(p0 + p1);
            P[r + hf * 8][ln]      = (bf16)p0;
            P[r + hf * 8][ln + 16] = (bf16)p1;
        }
        __syncthreads();
        // re-layout P: C-fragment -> A-fragment (16x32) via LDS
        v16bf pf;
        {
            const bf16* pp = &P[ln][hf * 16];
            v8bf lo = *(const v8bf*)pp;
            v8bf hi = *(const v8bf*)(pp + 8);
#pragma unroll
            for (int e = 0; e < 8; ++e) { pf[e] = lo[e]; pf[e + 8] = hi[e]; }
        }
        // O += P*V, V fragments in groups of 4
#pragma unroll
        for (int g = 0; g < 4; ++g) {
            v16bf vg[4];
#pragma unroll
            for (int t = 0; t < 4; ++t) {
                int ct = g * 4 + t;
                int m = ct >> 2;
                int feat = h * DD + (ct & 3) * 16 + ln;
                vg[t] = *(const v16bf*)(VpT + (((size_t)b * MM + m) * DVV + feat) * KNN
                                        + key0 + hf * 16);
            }
#pragma unroll
            for (int t = 0; t < 4; ++t)
                oacc[g * 4 + t] = wmma_bf16(pf, vg[t], oacc[g * 4 + t]);
        }
        __syncthreads();
    }

    float inv[8];
#pragma unroll
    for (int r = 0; r < 8; ++r) inv[r] = 1.0f / (lsum[r] + 1e-8f);
#pragma unroll
    for (int ct = 0; ct < 16; ++ct) {
        int m = ct >> 2;
        int feat = h * DD + (ct & 3) * 16 + ln;
#pragma unroll
        for (int r = 0; r < 8; ++r) {
            int row = b * NN + n0 + r + hf * 8;
            Obf[((size_t)row * MM + m) * DVV + feat] = (bf16)(oacc[ct][r] * inv[r]);
        }
    }
}

// ---------------------------------------------------------------------------
extern "C" void kernel_launch(void* const* d_in, const int* in_sizes, int n_in,
                              void* d_out, int out_size, void* d_ws, size_t ws_size,
                              hipStream_t stream) {
    const float* Q   = (const float*)d_in[0];
    const float* K   = (const float*)d_in[1];
    const int*   msk = (const int*)d_in[2];
    const float* w_q = (const float*)d_in[3];
    const float* w_k = (const float*)d_in[4];
    const float* w_v = (const float*)d_in[5];
    const float* w_o = (const float*)d_in[6];
    float* out = (float*)d_out;

    // workspace partition (bf16 elements)
    bf16* w = (bf16*)d_ws;
    size_t o = 0;
    bf16* Qbf = w + o; o += (size_t)ROWS * DQK;       // 16384*512
    bf16* Kbf = w + o; o += (size_t)ROWS * DQK;
    bf16* wqT = w + o; o += (size_t)DVV * DQK;        // [1024][512]
    bf16* wkT = w + o; o += (size_t)DVV * DQK;
    bf16* wvT = w + o; o += (size_t)DVV * DQK;
    bf16* woT = w + o; o += (size_t)DVV * DVV;        // [1024][1024]
    bf16* Qp  = w + o; o += (size_t)ROWS * DVV;       // 16384*1024
    bf16* Kp  = w + o; o += (size_t)ROWS * DVV;
    bf16* VpT = w + o; o += (size_t)ROWS * DVV;       // [b][m][1024][512]
    bf16* Obf = w + o; o += (size_t)ROWS * DVV;

    // 1) convert inputs to bf16
    {
        int n4 = (ROWS * DQK) / 4;
        cvt_f32_bf16_kernel<<<(n4 + 255) / 256, 256, 0, stream>>>(Q, Qbf, n4);
        cvt_f32_bf16_kernel<<<(n4 + 255) / 256, 256, 0, stream>>>(K, Kbf, n4);
    }
    // 2) transpose-convert weights
    transpose_cvt_kernel<<<dim3(DVV / 256, DQK), 256, 0, stream>>>(w_q, wqT, DQK, DVV);
    transpose_cvt_kernel<<<dim3(DVV / 256, DQK), 256, 0, stream>>>(w_k, wkT, DQK, DVV);
    transpose_cvt_kernel<<<dim3(DVV / 256, DQK), 256, 0, stream>>>(w_v, wvT, DQK, DVV);
    transpose_cvt_kernel<<<dim3(DVV / 256, DVV), 256, 0, stream>>>(w_o, woT, DVV, DVV);

    // 3) projection GEMMs: [16384,512] x [512,1024]
    {
        dim3 grid(ROWS / 128, DVV / 64);
        // fold 1/sqrt(DV)=1/32 into Qp (exact power of two, only consumed by attention)
        gemm_bf16_kernel<0><<<grid, 256, 0, stream>>>(Qbf, wqT, (void*)Qp, DQK, DVV, 0.03125f);
        gemm_bf16_kernel<0><<<grid, 256, 0, stream>>>(Kbf, wkT, (void*)Kp, DQK, DVV, 1.0f);
        gemm_bf16_kernel<2><<<grid, 256, 0, stream>>>(Kbf, wvT, (void*)VpT, DQK, DVV, 1.0f);
    }

    // 4) attention: grid (N/16, H, B), 1 wave per block
    attn_kernel<<<dim3(NN / 16, HH, BB), 32, 0, stream>>>(Qp, Kp, VpT, msk, Obf);

    // 5) output GEMM: [16384,1024] x [1024,1024] -> f32
    {
        dim3 grid(ROWS / 128, DVV / 64);
        gemm_bf16_kernel<1><<<grid, 256, 0, stream>>>(Obf, woT, (void*)out, DVV, DVV, 1.0f);
    }
}
